// GAT_AZ_Toys_14362370638273
// MI455X (gfx1250) — matrix operand order repaired and checked by reading, not compiled, not run on hardware
//
#include <hip/hip_runtime.h>
#include <hip/hip_bf16.h>
#include <cstddef>

#define EMB   32
#define H1    8
#define OUTD  32
#define NI    50000
#define NF    20000
#define NE    500000
#define BSZ   4096

typedef __attribute__((ext_vector_type(16))) __bf16 v16bf;
typedef __attribute__((ext_vector_type(8)))  __bf16 v8bf;
typedef __attribute__((ext_vector_type(8)))  float  v8f;

// ---------------------------------------------------------------- utilities
__global__ void f32_to_bf16_k(const float* __restrict__ x, __bf16* __restrict__ y, int n) {
  int i = blockIdx.x * blockDim.x + threadIdx.x;
  if (i < n) y[i] = (__bf16)x[i];
}

__global__ void init_bias3_k(float* __restrict__ out, int n, int C,
                             const float* __restrict__ b0, const float* __restrict__ b1,
                             const float* __restrict__ b2) {
  int i = blockIdx.x * blockDim.x + threadIdx.x;
  if (i >= n * C) return;
  int c = i % C;
  float v = 0.f;
  if (b0) v += b0[c];
  if (b1) v += b1[c];
  if (b2) v += b2[c];
  out[i] = v;
}

__global__ void build_item_in_k(const float* __restrict__ emb, const int* __restrict__ id,
                                const int* __restrict__ rating, const float* __restrict__ ts,
                                float* __restrict__ out, int n) {
  int i = blockIdx.x * blockDim.x + threadIdx.x;
  if (i >= n * 96) return;
  int node = i / 96, c = i - node * 96;
  float v;
  if (c < 32)      v = emb[id[node] * 32 + c];
  else if (c < 64) v = emb[rating[node] * 32 + (c - 32)];
  else             v = ts[node * 32 + (c - 64)];
  out[i] = v;
}

// ------------------------------------------------- WMMA GEMM (bf16 -> f32)
// Y[n,N] = rowgather(X)[n,K] @ Wb[K,N] (+ bias[N]).  K in {32,96,256} (templated,
// fully unrolled -> back-to-back v_wmma per wave).  Block = 4 waves (128 thr),
// 16-row x 64-col tile; wave w owns one 16x16 D-tile of V_WMMA_F32_16X16X32_BF16.
// LDS staging (rows padded to 80B => every lane fragment run is 16B-aligned):
//   As[r][k]      : A tile, row-major      -> a-frag = 2 aligned 16B runs/lane
//   Bs[c][k]      : B tile, TRANSPOSED     -> b-frag = 2 aligned 16B runs/lane
// Fragment layouts per CDNA5 ISA 7.12.2:
//   A: lane m=L&15, kg=L>>4; elems 0..7 = K[8kg..8kg+7], 8..15 = K[16+8kg..+7]
//   B: lane c=L&15, kg=L>>4; elems 0..7 = K[8kg..8kg+7], 8..15 = K[16+8kg..+7]
//   C/D: lane n=L&15; acc[r] is row m = r + 8*(L>>4)
template <int K>
__global__ __launch_bounds__(128) void gemm_wmma_bf16_k(
    const float* __restrict__ X, const int* __restrict__ gather,
    const __bf16* __restrict__ Wb, const float* __restrict__ bias,
    float* __restrict__ Y, int n, int N) {
  __shared__ __bf16 As[16][40];   // 80B rows
  __shared__ __bf16 Bs[64][40];   // transposed: Bs[col][k], 80B rows
  const int tid  = threadIdx.x;
  const int wave = tid >> 5, lane = tid & 31;
  const int m    = lane & 15;
  const int hi   = lane >> 4;
  const int rowBase = blockIdx.x * 16;
  const int colBase = blockIdx.y * 64;
  const int colTile = colBase + wave * 16;
  v8f acc = {};
#pragma unroll
  for (int k0 = 0; k0 < K; k0 += 32) {
#pragma unroll
    for (int i = tid; i < 16 * 32; i += 128) {       // stage A (gather + cvt)
      int r = i >> 5, k = i & 31;
      int row = rowBase + r;
      float v = 0.f;
      if (row < n) {
        int srow = gather ? gather[row] : row;
        v = X[srow * K + k0 + k];
      }
      As[r][k] = (__bf16)v;
    }
#pragma unroll
    for (int i = tid; i < 32 * 64; i += 128) {       // stage B (coalesced read,
      int k = i >> 6, c = i & 63;                    //  transposed LDS write)
      int col = colBase + c;
      Bs[c][k] = (col < N) ? Wb[(k0 + k) * N + col] : (__bf16)0.f;
    }
    __syncthreads();
    if (colTile < N) {
      v8bf alo = *(const v8bf*)&As[m][8 * hi];
      v8bf ahi = *(const v8bf*)&As[m][16 + 8 * hi];
      v8bf blo = *(const v8bf*)&Bs[wave * 16 + m][8 * hi];
      v8bf bhi = *(const v8bf*)&Bs[wave * 16 + m][16 + 8 * hi];
      v16bf a = __builtin_shufflevector(alo, ahi, 0,1,2,3,4,5,6,7,8,9,10,11,12,13,14,15);
      v16bf b = __builtin_shufflevector(blo, bhi, 0,1,2,3,4,5,6,7,8,9,10,11,12,13,14,15);
      acc = __builtin_amdgcn_wmma_f32_16x16x32_bf16(false, a, false, b,
                                                    (short)0, acc, false, false);
    }
    __syncthreads();
  }
  if (colTile < N) {
    const int col = colTile + m;
    const float bv = bias ? bias[col] : 0.f;
    float* yp = Y + (rowBase + 8 * hi) * N + col;
    if (rowBase + 16 <= n) {                          // full-tile fast path
#pragma unroll
      for (int r = 0; r < 8; ++r) yp[r * N] = acc[r] + bv;
    } else {
#pragma unroll
      for (int r = 0; r < 8; ++r)
        if (rowBase + 8 * hi + r < n) yp[r * N] = acc[r] + bv;
    }
  }
}

// ------------------------------------------- attention-logit helper kernels
// wfold[k,h] = sum_d W[k, h*D+d] * a[h,d]   (folds attention vec into W)
__global__ void fold_attn_k(const float* __restrict__ W, const float* __restrict__ a,
                            float* __restrict__ wfold, int K, int H, int D) {
  int i = blockIdx.x * blockDim.x + threadIdx.x;
  if (i >= K * H) return;
  int k = i / H, h = i - k * H;
  float s = 0.f;
  for (int d = 0; d < D; ++d) s += W[k * (H * D) + h * D + d] * a[h * D + d];
  wfold[i] = s;
}

// e[i,h] = sum_k X[i,k] * wfold[k,h]
__global__ void node_attn_k(const float* __restrict__ X, const float* __restrict__ wfold,
                            float* __restrict__ e, int n, int K, int H) {
  int i = blockIdx.x * blockDim.x + threadIdx.x;
  if (i >= n * H) return;
  int node = i / H, h = i - node * H;
  float s = 0.f;
  for (int k = 0; k < K; ++k) s += X[node * K + k] * wfold[k * H + h];
  e[i] = s;
}

// ------------------------------------------------------- edge softmax + agg
// Pass 1: ex = exp(leaky_relu(el[src]+er[dst])); segment-sum into ssum[dst].
// (segment-max dropped: softmax is shift-invariant; logits here are << 1.)
__global__ void edge_pass1_k(const int* __restrict__ src, const int* __restrict__ dst,
                             const float* __restrict__ el, const float* __restrict__ er,
                             float* __restrict__ ex, float* __restrict__ ssum,
                             int E, int H) {
  int i = blockIdx.x * blockDim.x + threadIdx.x;
  if (i >= E * H) return;
  int e = i / H, h = i - e * H;
  float v = el[src[e] * H + h] + er[dst[e] * H + h];
  v = (v > 0.f) ? v : 0.2f * v;
  float x = __expf(v);
  ex[i] = x;
  atomicAdd(&ssum[dst[e] * H + h], x);
}

// Pass 2 (bandwidth-critical): out[dst,c] += hs[src,c] * ex[e,c/D]/max(ssum,1e-9)
// float4 per thread: 1 global_load_b128 + 4 global_atomic_add_f32.
__global__ void edge_pass2_k(const int* __restrict__ src, const int* __restrict__ dst,
                             const float* __restrict__ ex, const float* __restrict__ ssum,
                             const float* __restrict__ hs, float* __restrict__ out,
                             int E, int H, int D) {
  const int C  = H * D;
  const int C4 = C >> 2;
  int i = blockIdx.x * blockDim.x + threadIdx.x;   // E*C4 <= 32M, fits int
  if (i >= E * C4) return;
  int e = i / C4, q = i - e * C4;
  int c = q << 2;
  int h = c / D;
  int s = src[e], d = dst[e];
  float a = ex[e * H + h] / fmaxf(ssum[d * H + h], 1e-9f);
  const float4 hv = *(const float4*)&hs[s * C + c];
  float* op = &out[d * C + c];
  atomicAdd(op + 0, hv.x * a);
  atomicAdd(op + 1, hv.y * a);
  atomicAdd(op + 2, hv.z * a);
  atomicAdd(op + 3, hv.w * a);
}

// ------------------------------------------------------------- scoring head
__global__ void final_score_k(const float* __restrict__ user_emb, const float* __restrict__ h2_item,
                              const int* __restrict__ tgt, float* __restrict__ out, int bs) {
  int b = blockIdx.x * blockDim.x + threadIdx.x;
  if (b >= bs) return;
  const float* u = user_emb + b * OUTD;
  const float* t = h2_item + tgt[b] * OUTD;
  float num = 0.f, nu = 0.f, nt = 0.f;
#pragma unroll 8
  for (int k = 0; k < OUTD; ++k) { num += u[k] * t[k]; nu += u[k] * u[k]; nt += t[k] * t[k]; }
  float den = fmaxf(sqrtf(nu) * sqrtf(nt), 1e-8f);
  float z = num / den;
  out[b] = 1.f / (1.f + __expf(-z));
}

// ---------------------------------------------------------------- host side
static inline int cdiv(long long a, long long b) { return (int)((a + b - 1) / b); }

extern "C" void kernel_launch(void* const* d_in, const int* in_sizes, int n_in,
                              void* d_out, int out_size, void* d_ws, size_t ws_size,
                              hipStream_t stream) {
  // ---- params (flattened in setup_inputs insertion order) ----
  const float* emb     = (const float*)d_in[0];
  const float* user_W  = (const float*)d_in[1];
  const float* user_b  = (const float*)d_in[2];
  const float* item_W  = (const float*)d_in[3];
  const float* item_b  = (const float*)d_in[4];
  const float* feat_W  = (const float*)d_in[5];
  const float* feat_b  = (const float*)d_in[6];
  const float* W1[4];  const float* al1[4]; const float* ar1[4]; const float* b1[4];
  for (int r = 0; r < 4; ++r) {
    W1[r]  = (const float*)d_in[7 + 4 * r + 0];
    al1[r] = (const float*)d_in[7 + 4 * r + 1];
    ar1[r] = (const float*)d_in[7 + 4 * r + 2];
    b1[r]  = (const float*)d_in[7 + 4 * r + 3];
  }
  const float* W2_hi  = (const float*)d_in[27];
  const float* al2_hi = (const float*)d_in[28];
  const float* ar2_hi = (const float*)d_in[29];
  const float* b2_hi  = (const float*)d_in[30];
  const float* item_ts    = (const float*)d_in[31];
  const int*   user_feat  = (const int*)d_in[32];
  const int*   item_id    = (const int*)d_in[33];
  const int*   item_rate  = (const int*)d_in[34];
  const int*   feat_id    = (const int*)d_in[35];
  const int* esrc[4] = { (const int*)d_in[36], (const int*)d_in[38],
                         (const int*)d_in[40], (const int*)d_in[42] };
  const int* edst[4] = { (const int*)d_in[37], (const int*)d_in[39],
                         (const int*)d_in[41], (const int*)d_in[43] };
  const int* target_idx = (const int*)d_in[44];
  float* out = (float*)d_out;

  // ---- workspace bump allocator ----
  size_t off = 0;
  auto alloc = [&](size_t bytes) -> void* {
    off = (off + 255) & ~(size_t)255;
    void* p = (char*)d_ws + off;
    off += bytes;
    return p;
  };
  __bf16* wb_user = (__bf16*)alloc(EMB * EMB * sizeof(__bf16));
  __bf16* wb_item = (__bf16*)alloc(3 * EMB * EMB * sizeof(__bf16));
  __bf16* wb_feat = (__bf16*)alloc(EMB * EMB * sizeof(__bf16));
  __bf16* wb_W1[4];
  for (int r = 0; r < 4; ++r) wb_W1[r] = (__bf16*)alloc(EMB * H1 * EMB * sizeof(__bf16));
  __bf16* wb_W2hi = (__bf16*)alloc(H1 * EMB * OUTD * sizeof(__bf16));
  float* wl  = (float*)alloc(256 * sizeof(float));
  float* wr  = (float*)alloc(256 * sizeof(float));
  float* hs       = (float*)alloc((size_t)NI * 256 * sizeof(float));  // also hosts item_in
  float* item_in  = hs;                                               // NI*96 <= NI*256
  float* item_emb = (float*)alloc((size_t)NI * EMB * sizeof(float));
  float* feat_emb = (float*)alloc((size_t)NF * EMB * sizeof(float));
  float* user_emb = (float*)alloc((size_t)BSZ * EMB * sizeof(float));
  float* el   = (float*)alloc((size_t)NI * H1 * sizeof(float));
  float* er   = (float*)alloc((size_t)NI * H1 * sizeof(float));
  float* ex   = (float*)alloc((size_t)NE * H1 * sizeof(float));
  float* ssum = (float*)alloc((size_t)NI * H1 * sizeof(float));
  float* h_item  = (float*)alloc((size_t)NI * 256 * sizeof(float));
  float* h_feat  = (float*)alloc((size_t)NF * 256 * sizeof(float));
  float* h2_item = (float*)alloc((size_t)NI * OUTD * sizeof(float));
  (void)ws_size; (void)n_in; (void)in_sizes; (void)out_size;

  auto cvt = [&](const float* src, __bf16* dst, int n) {
    f32_to_bf16_k<<<cdiv(n, 256), 256, 0, stream>>>(src, dst, n);
  };
  auto gemm = [&](const float* X, const int* g, const __bf16* Wb, const float* bias,
                  float* Y, int n, int K, int N) {
    dim3 grid(cdiv(n, 16), cdiv(N, 64));
    if (K == 32)
      gemm_wmma_bf16_k<32><<<grid, 128, 0, stream>>>(X, g, Wb, bias, Y, n, N);
    else if (K == 96)
      gemm_wmma_bf16_k<96><<<grid, 128, 0, stream>>>(X, g, Wb, bias, Y, n, N);
    else
      gemm_wmma_bf16_k<256><<<grid, 128, 0, stream>>>(X, g, Wb, bias, Y, n, N);
  };

  // ---- 0) weights -> bf16 ----
  cvt(user_W, wb_user, EMB * EMB);
  cvt(item_W, wb_item, 3 * EMB * EMB);
  cvt(feat_W, wb_feat, EMB * EMB);
  for (int r = 0; r < 4; ++r) cvt(W1[r], wb_W1[r], EMB * H1 * EMB);
  cvt(W2_hi, wb_W2hi, H1 * EMB * OUTD);

  // ---- 1) node embeddings (WMMA GEMMs) ----
  build_item_in_k<<<cdiv((long long)NI * 96, 256), 256, 0, stream>>>(
      emb, item_id, item_rate, item_ts, item_in, NI);
  gemm(item_in, nullptr, wb_item, item_b, item_emb, NI, 96, EMB);
  gemm(emb, feat_id, wb_feat, feat_b, feat_emb, NF, EMB, EMB);
  gemm(emb, user_feat, wb_user, user_b, user_emb, BSZ, EMB, EMB);

  // ---- 2) layer1: HeteroGraphConv, sum-aggregated per dst type ----
  init_bias3_k<<<cdiv((long long)NF * 256, 256), 256, 0, stream>>>(
      h_feat, NF, 256, b1[0], nullptr, nullptr);
  init_bias3_k<<<cdiv((long long)NI * 256, 256), 256, 0, stream>>>(
      h_item, NI, 256, b1[1], b1[2], b1[3]);

  struct Rel { const float* xs; int ns; const float* xd; int nd; int e; float* acc; };
  const Rel rels[4] = {
      { item_emb, NI, feat_emb, NF, 0, h_feat },   // belongto
      { feat_emb, NF, item_emb, NI, 1, h_item },   // hasinstance
      { item_emb, NI, item_emb, NI, 2, h_item },   // interacted
      { item_emb, NI, item_emb, NI, 3, h_item },   // clickby
  };
  for (int r = 0; r < 4; ++r) {
    const Rel& R = rels[r];
    fold_attn_k<<<1, 256, 0, stream>>>(W1[r], al1[r], wl, EMB, H1, EMB);
    fold_attn_k<<<1, 256, 0, stream>>>(W1[r], ar1[r], wr, EMB, H1, EMB);
    node_attn_k<<<cdiv((long long)R.ns * H1, 256), 256, 0, stream>>>(R.xs, wl, el, R.ns, EMB, H1);
    node_attn_k<<<cdiv((long long)R.nd * H1, 256), 256, 0, stream>>>(R.xd, wr, er, R.nd, EMB, H1);
    gemm(R.xs, nullptr, wb_W1[r], nullptr, hs, R.ns, EMB, H1 * EMB);
    hipMemsetAsync(ssum, 0, (size_t)R.nd * H1 * sizeof(float), stream);
    edge_pass1_k<<<cdiv((long long)NE * H1, 256), 256, 0, stream>>>(
        esrc[R.e], edst[R.e], el, er, ex, ssum, NE, H1);
    edge_pass2_k<<<cdiv((long long)NE * 64, 256), 256, 0, stream>>>(
        esrc[R.e], edst[R.e], ex, ssum, hs, R.acc, NE, H1, EMB);
  }

  // ---- 3) layer2: only 'hasinstance' feeds h2_item; h2_feat is dead code ----
  fold_attn_k<<<1, 256, 0, stream>>>(W2_hi, al2_hi, wl, H1 * EMB, 1, OUTD);
  fold_attn_k<<<1, 256, 0, stream>>>(W2_hi, ar2_hi, wr, H1 * EMB, 1, OUTD);
  node_attn_k<<<cdiv(NF, 256), 256, 0, stream>>>(h_feat, wl, el, NF, H1 * EMB, 1);
  node_attn_k<<<cdiv(NI, 256), 256, 0, stream>>>(h_item, wr, er, NI, H1 * EMB, 1);
  gemm(h_feat, nullptr, wb_W2hi, nullptr, hs, NF, H1 * EMB, OUTD);  // hs2 [NF,32]
  init_bias3_k<<<cdiv((long long)NI * OUTD, 256), 256, 0, stream>>>(
      h2_item, NI, OUTD, b2_hi, nullptr, nullptr);
  hipMemsetAsync(ssum, 0, (size_t)NI * sizeof(float), stream);
  edge_pass1_k<<<cdiv((long long)NE, 256), 256, 0, stream>>>(
      esrc[1], edst[1], el, er, ex, ssum, NE, 1);
  edge_pass2_k<<<cdiv((long long)NE * 8, 256), 256, 0, stream>>>(
      esrc[1], edst[1], ex, ssum, hs, h2_item, NE, 1, OUTD);

  // ---- 4) cosine-sim + sigmoid head ----
  final_score_k<<<cdiv(BSZ, 256), 256, 0, stream>>>(user_emb, h2_item, target_idx, out, BSZ);
}
